// Attention_35124242547322
// MI455X (gfx1250) — compile-verified, matrix operand order checked by reading, not run on hardware
//
#include <hip/hip_runtime.h>
#include <hip/hip_bf16.h>
#include <math.h>

// Sizes (fixed by the reference)
#define LL 16384
#define DD 1024
#define AA 1024

typedef __attribute__((ext_vector_type(16))) __bf16 v16bf;
typedef __attribute__((ext_vector_type(8)))  float  v8f;

// Gather one 16x16x32 bf16 WMMA operand fragment for this lane.
// Per CDNA5 ISA (16-bit A 16x32 layout, B mirrors with lane = column):
//   lane holds row/col (lane & 15); K base = ((lane>>4)<<3) already folded
//   into p by the caller. Elements 0..7  <- p[0..7]   (K = kb+0..7)
//   Elements 8..15 <- p[16..23] (K = kb+16..23)
__device__ __forceinline__ v16bf load_frag_bf16(const float* __restrict__ p) {
    float4 x0 = *(const float4*)(p + 0);
    float4 x1 = *(const float4*)(p + 4);
    float4 y0 = *(const float4*)(p + 16);
    float4 y1 = *(const float4*)(p + 20);
    v16bf v;
    v[0]  = (__bf16)x0.x; v[1]  = (__bf16)x0.y; v[2]  = (__bf16)x0.z; v[3]  = (__bf16)x0.w;
    v[4]  = (__bf16)x1.x; v[5]  = (__bf16)x1.y; v[6]  = (__bf16)x1.z; v[7]  = (__bf16)x1.w;
    v[8]  = (__bf16)y0.x; v[9]  = (__bf16)y0.y; v[10] = (__bf16)y0.z; v[11] = (__bf16)y0.w;
    v[12] = (__bf16)y1.x; v[13] = (__bf16)y1.y; v[14] = (__bf16)y1.z; v[15] = (__bf16)y1.w;
    return v;
}

// ---------------------------------------------------------------------------
// Kernel 1: c[a] = dot(W_att[a, D:2D], h_t) + b_att[a]
// ---------------------------------------------------------------------------
__global__ void prep_c_kernel(const float* __restrict__ W,
                              const float* __restrict__ ht,
                              const float* __restrict__ b,
                              float* __restrict__ cvec) {
    int a = blockIdx.x * blockDim.x + threadIdx.x;      // 0..AA-1
    const float4* wr = (const float4*)(W + (size_t)a * (2 * DD) + DD);
    const float4* h4 = (const float4*)ht;
    float s = 0.f;
    for (int k = 0; k < DD / 4; ++k) {
        float4 wv = wr[k];
        float4 hv = h4[k];
        s += wv.x * hv.x + wv.y * hv.y + wv.z * hv.z + wv.w * hv.w;
    }
    cvec[a] = s + b[a];
}

// ---------------------------------------------------------------------------
// Kernel 2: fused  beta[l] = sum_a tanh( h_i[l]·W1[a] + c[a] ) * u[a]
// Block: 32 rows of h_i x all 1024 columns. 8 waves, each wave a 128-col slice.
// bf16 WMMA, f32 accumulate. Epilogue reduces beta via LDS float atomics.
// ---------------------------------------------------------------------------
__global__ __launch_bounds__(256, 1)
void gemm_tanh_beta_kernel(const float* __restrict__ hi,
                           const float* __restrict__ W,
                           const float* __restrict__ cvec,
                           const float* __restrict__ u,
                           float* __restrict__ beta) {
    __shared__ float beta_lds[32];

    const int tid    = threadIdx.x;
    const int lane   = tid & 31;
    const int wave   = tid >> 5;
    const int lane16 = lane & 15;
    const int half8  = (lane >> 4) << 3;   // 0 or 8 : K sub-offset & M sub-offset
    const int m0     = blockIdx.x * 32;
    const int n_base = wave * 128;

    const float* arow0 = hi + (size_t)(m0 + lane16) * DD;
    const float* arow1 = arow0 + (size_t)16 * DD;

    v8f zero8 = {0.f, 0.f, 0.f, 0.f, 0.f, 0.f, 0.f, 0.f};
    v8f acc[2][8];
#pragma unroll
    for (int mi = 0; mi < 2; ++mi)
#pragma unroll
        for (int j = 0; j < 8; ++j) acc[mi][j] = zero8;

    for (int k0 = 0; k0 < DD; k0 += 32) {
        const int kb = k0 + half8;
        v16bf a0 = load_frag_bf16(arow0 + kb);
        v16bf a1 = load_frag_bf16(arow1 + kb);
#pragma unroll
        for (int j = 0; j < 8; ++j) {
            const float* brow = W + (size_t)(n_base + j * 16 + lane16) * (2 * DD) + kb;
            v16bf bf = load_frag_bf16(brow);
            acc[0][j] = __builtin_amdgcn_wmma_f32_16x16x32_bf16(
                false, a0, false, bf, (short)0, acc[0][j], false, false);
            acc[1][j] = __builtin_amdgcn_wmma_f32_16x16x32_bf16(
                false, a1, false, bf, (short)0, acc[1][j], false, false);
        }
    }

    // Epilogue: C layout => VGPR r on this lane holds (M = r + half8, N = lane16)
    float bsum[2][8];
#pragma unroll
    for (int mi = 0; mi < 2; ++mi)
#pragma unroll
        for (int r = 0; r < 8; ++r) bsum[mi][r] = 0.f;

#pragma unroll
    for (int j = 0; j < 8; ++j) {
        const int n = n_base + j * 16 + lane16;
        const float cj = cvec[n];
        const float uj = u[n];
#pragma unroll
        for (int mi = 0; mi < 2; ++mi)
#pragma unroll
            for (int r = 0; r < 8; ++r)
                bsum[mi][r] += tanhf(acc[mi][j][r] + cj) * uj;
    }

    if (tid < 32) beta_lds[tid] = 0.f;
    __syncthreads();
#pragma unroll
    for (int mi = 0; mi < 2; ++mi)
#pragma unroll
        for (int r = 0; r < 8; ++r)
            atomicAdd(&beta_lds[mi * 16 + half8 + r], bsum[mi][r]);
    __syncthreads();
    if (tid < 32) beta[m0 + tid] = beta_lds[tid];
}

// ---------------------------------------------------------------------------
// Kernel 3: softmax weights  w[l] = exp(beta[l]-max);  scal = 1/sum
// ---------------------------------------------------------------------------
__global__ void softmax_kernel(const float* __restrict__ beta,
                               float* __restrict__ w,
                               float* __restrict__ scal) {
    __shared__ float red[1024];
    const int tid = threadIdx.x;

    float m = -3.402823466e38f;
    for (int i = tid; i < LL; i += 1024) m = fmaxf(m, beta[i]);
    red[tid] = m;
    __syncthreads();
    for (int s = 512; s > 0; s >>= 1) {
        if (tid < s) red[tid] = fmaxf(red[tid], red[tid + s]);
        __syncthreads();
    }
    const float M = red[0];
    __syncthreads();

    float sum = 0.f;
    for (int i = tid; i < LL; i += 1024) {
        float e = expf(beta[i] - M);
        w[i] = e;
        sum += e;
    }
    red[tid] = sum;
    __syncthreads();
    for (int s = 512; s > 0; s >>= 1) {
        if (tid < s) red[tid] += red[tid + s];
        __syncthreads();
    }
    if (tid == 0) scal[0] = 1.0f / red[0];
}

// ---------------------------------------------------------------------------
// Kernel 4: partial weighted sums over L chunks of 256 rows (coalesced f4)
// ---------------------------------------------------------------------------
__global__ __launch_bounds__(256)
void wsum_partial_kernel(const float* __restrict__ hi,
                         const float* __restrict__ w,
                         float* __restrict__ partial) {
    __shared__ float lw[256];
    const int tid = threadIdx.x;
    const int l0 = blockIdx.x * 256;
    lw[tid] = w[l0 + tid];
    __syncthreads();

    const float4* h4 = (const float4*)hi;
    float4 acc = {0.f, 0.f, 0.f, 0.f};
    for (int i = 0; i < 256; ++i) {
        const float wv = lw[i];
        float4 h = h4[(size_t)(l0 + i) * (DD / 4) + tid];
        acc.x += wv * h.x;
        acc.y += wv * h.y;
        acc.z += wv * h.z;
        acc.w += wv * h.w;
    }
    ((float4*)partial)[(size_t)blockIdx.x * (DD / 4) + tid] = acc;
}

// ---------------------------------------------------------------------------
// Kernel 5: reduce partials over the 64 L-chunks and scale by 1/sum
// ---------------------------------------------------------------------------
__global__ void wsum_reduce_kernel(const float* __restrict__ partial,
                                   const float* __restrict__ scal,
                                   float* __restrict__ out) {
    const int d = blockIdx.x * blockDim.x + threadIdx.x;   // 0..DD-1
    float s = 0.f;
    for (int c = 0; c < LL / 256; ++c) s += partial[(size_t)c * DD + d];
    out[d] = s * scal[0];
}

// ---------------------------------------------------------------------------
// Workspace layout (bytes):
//   beta   @ 0        : 16384 * 4 = 65536
//   cvec   @ 65536    :  1024 * 4 =  4096
//   w      @ 69632    : 16384 * 4 = 65536
//   scal   @ 135168   : 256 (only 4 used)
//   partial@ 135424   : 64 * 1024 * 4 = 262144   (16B aligned)
// Total ~398 KB.
// ---------------------------------------------------------------------------
extern "C" void kernel_launch(void* const* d_in, const int* in_sizes, int n_in,
                              void* d_out, int out_size, void* d_ws, size_t ws_size,
                              hipStream_t stream) {
    const float* h_i   = (const float*)d_in[0];
    const float* h_t   = (const float*)d_in[1];
    const float* W_att = (const float*)d_in[2];
    const float* b_att = (const float*)d_in[3];
    const float* u     = (const float*)d_in[4];
    float* out = (float*)d_out;

    char* ws = (char*)d_ws;
    float* beta = (float*)(ws + 0);
    float* cvec = (float*)(ws + 65536);
    float* w    = (float*)(ws + 69632);
    float* scal = (float*)(ws + 135168);
    float* part = (float*)(ws + 135424);

    prep_c_kernel<<<AA / 256, 256, 0, stream>>>(W_att, h_t, b_att, cvec);
    gemm_tanh_beta_kernel<<<LL / 32, 256, 0, stream>>>(h_i, W_att, cvec, u, beta);
    softmax_kernel<<<1, 1024, 0, stream>>>(beta, w, scal);
    wsum_partial_kernel<<<LL / 256, 256, 0, stream>>>(h_i, w, part);
    wsum_reduce_kernel<<<DD / 256, 256, 0, stream>>>(part, scal, out);
}
